// ClassificationRNN_53876069761363
// MI455X (gfx1250) — compile-verified, matrix operand order checked by reading, not run on hardware
//
#include <hip/hip_runtime.h>
#include <hip/hip_bf16.h>
#include <stdint.h>

// ---------------------------------------------------------------------------
// ClassificationRNN on MI455X (gfx1250, wave32, WMMA)
//   B=512 T=300 V=50000 D=128 H=256 DENSE=256 C=14
// Pipeline:
//   k_f2bf / k_wcat / k_bias : one-time bf16 conversion + weight fusion
//   k_lstm  : async-to-LDS embed gather + [x|h]·[W_ih|W_hh]^T (bf16 WMMA,
//             K=384), in-register gates + cell state, 32 WGs x 16 rows
//   k_dense : [512,76800]x[76800,256] bf16 WMMA, 2x2 register tiling
//   k_head  : [512,256]x[256,14] + softmax (one wave per row)
// ---------------------------------------------------------------------------

typedef __attribute__((ext_vector_type(16))) __bf16 v16bf;
typedef __attribute__((ext_vector_type(8)))  float  v8f;

#define B_     512
#define T_     300
#define D_     128
#define H_     256
#define G4_    1024          // 4*H
#define KCAT   384           // D + H
#define DENSE_ 256
#define C_     14
#define KD     76800         // T*H

// workspace layout (bytes, 256B aligned)
#define OFF_EMB   ((size_t)0)           // 50001*128 bf16   = 12,800,256
#define OFF_WCAT  ((size_t)12800256)    // 1024*384 bf16    =    786,432
#define OFF_W3    ((size_t)13586688)    // 256*76800 bf16   = 39,321,600
#define OFF_BIAS  ((size_t)52908288)    // 1024 f32         =      4,096
#define OFF_HS    ((size_t)52912384)    // 512*300*256 bf16 = 78,643,200
#define OFF_OUT1  ((size_t)131555584)   // 512*256 f32      =    524,288

static __device__ __forceinline__ unsigned short f2bf(float f) {
    unsigned int u = __float_as_uint(f);
    u += 0x7fffu + ((u >> 16) & 1u);          // round-to-nearest-even
    return (unsigned short)(u >> 16);
}
static __device__ __forceinline__ float sigm(float x) {
    return 1.f / (1.f + __expf(-x));
}
static __device__ __forceinline__ float tanh_fast(float x) {
    return 2.f / (1.f + __expf(-2.f * x)) - 1.f;
}

// ---------------------------------------------------------------- converters
__global__ void k_f2bf(const float* __restrict__ s, unsigned short* __restrict__ d, int n) {
    for (int i = blockIdx.x * blockDim.x + threadIdx.x; i < n; i += gridDim.x * blockDim.x)
        d[i] = f2bf(s[i]);
}

// fused weight: row g = [ W_ih[g,0:128] | W_hh[g,0:256] ]  -> [1024,384] bf16
__global__ void k_wcat(const float* __restrict__ wih, const float* __restrict__ whh,
                       unsigned short* __restrict__ d) {
    const int n = G4_ * KCAT;
    for (int i = blockIdx.x * blockDim.x + threadIdx.x; i < n; i += gridDim.x * blockDim.x) {
        int g = i / KCAT, k = i - g * KCAT;
        float v = (k < D_) ? wih[g * D_ + k] : whh[g * H_ + (k - D_)];
        d[i] = f2bf(v);
    }
}

__global__ void k_bias(const float* __restrict__ a, const float* __restrict__ b,
                       float* __restrict__ d) {
    int i = blockIdx.x * blockDim.x + threadIdx.x;
    if (i < G4_) d[i] = a[i] + b[i];
}

// ---------------------------------------------------------------- LSTM core
// grid = 32 blocks (16 batch rows each), block = 128 threads (4 waves).
// Each wave owns 4 h-column tiles (n0 = (wave*4+q)*16) and the matching cell
// state in registers; it computes the i,f,g,o tiles for those columns.
__global__ void __launch_bounds__(128)
k_lstm(const int* __restrict__ tokens,             // [B,T]
       const unsigned short* __restrict__ embbf,   // [V+1,128] bf16
       const unsigned short* __restrict__ wcat,    // [1024,384] bf16
       const float* __restrict__ bias,             // [1024] = b_ih+b_hh
       unsigned short* __restrict__ hs)            // [B,T,H] bf16
{
    const int tid  = threadIdx.x;
    const int lane = tid & 31;
    const int wave = tid >> 5;          // 0..3
    const int lo   = lane & 15;
    const int hi   = lane >> 4;
    const int b0   = blockIdx.x * 16;

    __shared__ __align__(32) unsigned short xh[16][KCAT];   // [row][ x(128) | h(256) ]

    // zero the h region (h_{-1} = 0)
    for (int i = tid; i < 16 * H_; i += 128)
        xh[i >> 8][D_ + (i & 255)] = 0;

    v8f cst[4];
#pragma unroll
    for (int q = 0; q < 4; ++q) cst[q] = (v8f){};

    const int grow  = tid >> 3;   // gather row 0..15 (8 threads per row)
    const int gpart = tid & 7;    // 32 bytes per thread

    unsigned short hbf[4][8];

    for (int t = 0; t < T_; ++t) {
        // ---- async embedding gather: global -> LDS, tracked by ASYNCcnt ----
        {
            int tok = tokens[(b0 + grow) * T_ + t];
            const unsigned short* gp = embbf + (size_t)tok * D_ + gpart * 16;
            // generic pointer to __shared__: low 32 bits carry the LDS offset
            unsigned lds = (unsigned)(size_t)&xh[grow][gpart * 16];
            asm volatile(
                "global_load_async_to_lds_b128 %0, %1, off\n\t"
                "global_load_async_to_lds_b128 %0, %1, off offset:16"
                :: "v"(lds), "v"(gp) : "memory");
        }
        asm volatile("s_wait_asynccnt 0x0" ::: "memory");
        __syncthreads();   // x_t and h_{t-1} visible to all waves

        // ---- gates = [x|h] . Wcat^T + (b_ih+b_hh), then state update ----
#pragma unroll
        for (int q = 0; q < 4; ++q) {
            const int n0 = (wave * 4 + q) * 16;
            v8f acc[4];
#pragma unroll
            for (int g = 0; g < 4; ++g) {
                float bv = bias[g * H_ + n0 + lo];
#pragma unroll
                for (int r = 0; r < 8; ++r) acc[g][r] = bv;
            }
#pragma unroll 1
            for (int k = 0; k < KCAT / 32; ++k) {
                v16bf a = *(const v16bf*)&xh[lo][k * 32 + hi * 16];
#pragma unroll
                for (int g = 0; g < 4; ++g) {
                    const unsigned short* bp =
                        wcat + (size_t)(g * H_ + n0 + lo) * KCAT + k * 32 + hi * 16;
                    v16bf b = *(const v16bf*)bp;
                    acc[g] = __builtin_amdgcn_wmma_f32_16x16x32_bf16(
                        false, a, false, b, (short)0, acc[g], false, false);
                }
            }
            // i,f,o sigmoid; g tanh; c = f*c + i*g; h = o*tanh(c)
#pragma unroll
            for (int r = 0; r < 8; ++r) {
                float ig = sigm(acc[0][r]);
                float fg = sigm(acc[1][r]);
                float gg = tanh_fast(acc[2][r]);
                float og = sigm(acc[3][r]);
                float c  = fg * cst[q][r] + ig * gg;
                cst[q][r] = c;
                float h  = og * tanh_fast(c);
                hbf[q][r] = f2bf(h);
            }
        }
        __syncthreads();   // all reads of xh for this step are done

        // ---- write h_t back to LDS and to global hs[b,t,:] ----
#pragma unroll
        for (int q = 0; q < 4; ++q) {
            const int n0 = (wave * 4 + q) * 16;
#pragma unroll
            for (int r = 0; r < 8; ++r) {
                int m = r + 8 * hi;         // C-tile row (batch row in chunk)
                int n = n0 + lo;            // h column
                xh[m][D_ + n] = hbf[q][r];
                hs[((size_t)(b0 + m) * T_ + t) * H_ + n] = hbf[q][r];
            }
        }
        // next iteration's gather + barrier orders these writes
    }
}

// ---------------------------------------------------------------- dense layer
// out1[512,256] = relu(hs[512,76800] . W3^T + b3), bf16 WMMA, K=76800
// 2x2 register tiling: each wave owns 2 M-tiles x 2 N-tiles (4 accumulators),
// so each K-step is 4 loads -> 4 WMMAs and W3 L2 traffic is halved.
__global__ void __launch_bounds__(256)
k_dense(const unsigned short* __restrict__ hsb,    // [512,76800] bf16
        const unsigned short* __restrict__ w3b,    // [256,76800] bf16
        const float* __restrict__ b3,
        float* __restrict__ out1)                  // [512,256]
{
    const int lane = threadIdx.x & 31;
    const int wave = threadIdx.x >> 5;   // 0..7, each owns 2 N-tiles
    const int lo = lane & 15, hi = lane >> 4;
    const int m0 = blockIdx.x * 32;      // 16 blocks x 32 rows
    const int n0 = wave * 32;

    v8f acc00, acc01, acc10, acc11;      // [m-tile][n-tile]
    {
        float bv0 = b3[n0 + lo], bv1 = b3[n0 + 16 + lo];
#pragma unroll
        for (int r = 0; r < 8; ++r) {
            acc00[r] = bv0; acc01[r] = bv1;
            acc10[r] = bv0; acc11[r] = bv1;
        }
    }

    const unsigned short* ap0 = hsb + (size_t)(m0 + lo) * KD + hi * 16;
    const unsigned short* ap1 = ap0 + (size_t)16 * KD;
    const unsigned short* bp0 = w3b + (size_t)(n0 + lo) * KD + hi * 16;
    const unsigned short* bp1 = w3b + (size_t)(n0 + 16 + lo) * KD + hi * 16;

#pragma unroll 2
    for (int k = 0; k < KD / 32; ++k) {
        v16bf a0 = *(const v16bf*)(ap0 + k * 32);
        v16bf a1 = *(const v16bf*)(ap1 + k * 32);
        v16bf b0 = *(const v16bf*)(bp0 + k * 32);
        v16bf b1 = *(const v16bf*)(bp1 + k * 32);
        acc00 = __builtin_amdgcn_wmma_f32_16x16x32_bf16(false, a0, false, b0,
                                                        (short)0, acc00, false, false);
        acc01 = __builtin_amdgcn_wmma_f32_16x16x32_bf16(false, a0, false, b1,
                                                        (short)0, acc01, false, false);
        acc10 = __builtin_amdgcn_wmma_f32_16x16x32_bf16(false, a1, false, b0,
                                                        (short)0, acc10, false, false);
        acc11 = __builtin_amdgcn_wmma_f32_16x16x32_bf16(false, a1, false, b1,
                                                        (short)0, acc11, false, false);
        // stream-ahead prefetch (emits global_prefetch_b8); speculative, safe OOB
        __builtin_prefetch(ap0 + k * 32 + 4096, 0, 1);
        __builtin_prefetch(ap1 + k * 32 + 4096, 0, 1);
        __builtin_prefetch(bp0 + k * 32 + 4096, 0, 1);
        __builtin_prefetch(bp1 + k * 32 + 4096, 0, 1);
    }

#pragma unroll
    for (int r = 0; r < 8; ++r) {
        int ma = m0 + r + 8 * hi;
        int mb = ma + 16;
        out1[(size_t)ma * DENSE_ + n0 + lo]      = fmaxf(acc00[r], 0.f);
        out1[(size_t)ma * DENSE_ + n0 + 16 + lo] = fmaxf(acc01[r], 0.f);
        out1[(size_t)mb * DENSE_ + n0 + lo]      = fmaxf(acc10[r], 0.f);
        out1[(size_t)mb * DENSE_ + n0 + 16 + lo] = fmaxf(acc11[r], 0.f);
    }
}

// ---------------------------------------------------------------- head+softmax
__global__ void __launch_bounds__(32)
k_head(const float* __restrict__ out1,   // [512,256]
       const float* __restrict__ W4,     // [14,256]
       const float* __restrict__ b4,     // [14]
       float* __restrict__ out)          // [512,14]
{
    const int b    = blockIdx.x;
    const int lane = threadIdx.x;
    __shared__ float row[DENSE_];
    for (int i = lane; i < DENSE_; i += 32) row[i] = out1[(size_t)b * DENSE_ + i];
    __syncthreads();

    float logit = 0.f;
    if (lane < C_) {
        const float* w = W4 + (size_t)lane * DENSE_;
        float s = b4[lane];
#pragma unroll 4
        for (int i = 0; i < DENSE_; ++i) s = fmaf(row[i], w[i], s);
        logit = s;
    }
    float v = (lane < C_) ? logit : -3.4e38f;
#pragma unroll
    for (int m = 16; m >= 1; m >>= 1) v = fmaxf(v, __shfl_xor(v, m, 32));
    float e = (lane < C_) ? __expf(logit - v) : 0.f;
    float s = e;
#pragma unroll
    for (int m = 16; m >= 1; m >>= 1) s += __shfl_xor(s, m, 32);
    if (lane < C_) out[(size_t)b * C_ + lane] = e / s;
}

// ---------------------------------------------------------------- launcher
extern "C" void kernel_launch(void* const* d_in, const int* in_sizes, int n_in,
                              void* d_out, int out_size, void* d_ws, size_t ws_size,
                              hipStream_t stream) {
    const int*   tokens = (const int*)  d_in[0];   // inputs_arrays [B,T]
    // d_in[1] = traj_lens (unused by the reference computation)
    const float* emb    = (const float*)d_in[2];   // [V+1,128]
    const float* W_ih   = (const float*)d_in[3];   // [1024,128]
    const float* W_hh   = (const float*)d_in[4];   // [1024,256]
    const float* b_ih   = (const float*)d_in[5];   // [1024]
    const float* b_hh   = (const float*)d_in[6];   // [1024]
    const float* W3     = (const float*)d_in[7];   // [256,76800]
    const float* b3     = (const float*)d_in[8];   // [256]
    const float* W4     = (const float*)d_in[9];   // [14,256]
    const float* b4     = (const float*)d_in[10];  // [14]
    float* out = (float*)d_out;

    char* ws = (char*)d_ws;
    unsigned short* embbf = (unsigned short*)(ws + OFF_EMB);
    unsigned short* wcat  = (unsigned short*)(ws + OFF_WCAT);
    unsigned short* w3b   = (unsigned short*)(ws + OFF_W3);
    float*          bias  = (float*)         (ws + OFF_BIAS);
    unsigned short* hsb   = (unsigned short*)(ws + OFF_HS);
    float*          out1  = (float*)         (ws + OFF_OUT1);

    // one-time bf16 conversions (cheap: ~130MB streamed)
    k_f2bf<<<2048, 256, 0, stream>>>(emb, embbf, 50001 * D_);
    k_wcat<<<1024, 256, 0, stream>>>(W_ih, W_hh, wcat);
    k_f2bf<<<4096, 256, 0, stream>>>(W3, w3b, DENSE_ * KD);
    k_bias<<<4, 256, 0, stream>>>(b_ih, b_hh, bias);

    // fused embedding + LSTM recurrence (batch-parallel, no cross-WG sync)
    k_lstm<<<32, 128, 0, stream>>>(tokens, embbf, wcat, bias, hsb);

    // dense + relu
    k_dense<<<16, 256, 0, stream>>>(hsb, w3b, b3, out1);

    // classifier head + softmax
    k_head<<<512, 32, 0, stream>>>(out1, W4, b4, out);
}
